// GPT_36189394437028
// MI455X (gfx1250) — matrix operand order repaired and checked by reading, not compiled, not run on hardware
//
#include <hip/hip_runtime.h>
#include <math.h>

// ---------------- model dims ----------------
#define VOCAB 50257
#define TT    128
#define EE    256
#define HH    8
#define DD    256
#define LL    6
#define BB    16
#define HD    2048          // H*D
#define MM    2048          // B*T rows
#define NPAD_LM 50432       // vocab padded to multiple of 64 (and 256)

typedef __attribute__((ext_vector_type(16))) __bf16 v16bf;
typedef __attribute__((ext_vector_type(8)))  __bf16 v8bf;
typedef __attribute__((ext_vector_type(8)))  float  v8f;

union ABu { v16bf v; v8bf h[2]; };

// =====================================================================
// Embedding: h[row] = wte[x[row]] + wpe[row % T]
// =====================================================================
__global__ __launch_bounds__(256) void embed_kernel(const int* __restrict__ x,
                                                    const float* __restrict__ wte,
                                                    const float* __restrict__ wpe,
                                                    float* __restrict__ h) {
  int row = blockIdx.x;
  int e   = threadIdx.x;
  int tok = x[row];
  h[(size_t)row * EE + e] = wte[(size_t)tok * EE + e] + wpe[(size_t)(row & (TT - 1)) * EE + e];
}

// =====================================================================
// LayerNorm (population var), fp32 in -> bf16 out.  One wave per row.
// =====================================================================
__global__ __launch_bounds__(256) void ln_kernel(const float* __restrict__ in,
                                                 const float* __restrict__ g,
                                                 const float* __restrict__ b,
                                                 __bf16* __restrict__ out) {
  int w = threadIdx.x >> 5, lane = threadIdx.x & 31;
  int row = blockIdx.x * 8 + w;
  const float* x = in + (size_t)row * EE;
  float vals[8]; float s = 0.f, s2 = 0.f;
#pragma unroll
  for (int i = 0; i < 8; i++) { float t = x[lane * 8 + i]; vals[i] = t; s += t; s2 += t * t; }
#pragma unroll
  for (int m = 16; m >= 1; m >>= 1) { s += __shfl_xor(s, m, 32); s2 += __shfl_xor(s2, m, 32); }
  float mean = s * (1.f / EE);
  float var  = s2 * (1.f / EE) - mean * mean;
  float inv  = rsqrtf(var + 1e-5f);
#pragma unroll
  for (int i = 0; i < 8; i++) {
    int e = lane * 8 + i;
    out[(size_t)row * EE + e] = (__bf16)((vals[i] - mean) * inv * g[e] + b[e]);
  }
}

// =====================================================================
// Pack fp32 weight [K x N] (or its transpose) into WMMA B-fragment order:
// block(nt,kb) of 512 bf16: lane = (khalf<<4)|nlocal holds 16 contiguous
// bf16 = B[kb*32 + khalf*16 + j][nt*16 + nlocal].
// =====================================================================
__global__ __launch_bounds__(256) void pack_kernel(const float* __restrict__ W,
                                                   __bf16* __restrict__ out,
                                                   int K, int N, int Npad, int trans) {
  size_t idx = (size_t)blockIdx.x * 256 + threadIdx.x;
  size_t total = (size_t)Npad * (size_t)K;
  if (idx >= total) return;
  int KB = K >> 5;
  size_t blk = idx >> 9;
  int r = (int)(idx & 511);
  int lane = r >> 4, j = r & 15;
  size_t nt = blk / (size_t)KB;
  int kb = (int)(blk % (size_t)KB);
  int n  = (int)nt * 16 + (lane & 15);
  int kk = kb * 32 + (lane >> 4) * 16 + j;
  float val = 0.f;
  if (n < N) val = trans ? W[(size_t)n * K + kk] : W[(size_t)kk * N + n];
  out[idx] = (__bf16)val;
}

// =====================================================================
// GEMM fragment helpers
// =====================================================================
__device__ __forceinline__ void load_frags(const __bf16* ar0, const __bf16* ar1,
                                           const __bf16* bb, int KB, int kb, int hi,
                                           ABu a[2], v16bf b[4]) {
  int k0 = kb * 32;
  a[0].h[0] = *(const v8bf*)(ar0 + k0 + hi * 16);
  a[0].h[1] = *(const v8bf*)(ar0 + k0 + hi * 16 + 8);
  a[1].h[0] = *(const v8bf*)(ar1 + k0 + hi * 16);
  a[1].h[1] = *(const v8bf*)(ar1 + k0 + hi * 16 + 8);
#pragma unroll
  for (int t = 0; t < 4; t++)
    b[t] = *(const v16bf*)(bb + ((size_t)t * KB + kb) * 512);
}

__device__ __forceinline__ void wmma_all(v8f acc[2][4], ABu a[2], v16bf b[4]) {
#pragma unroll
  for (int m = 0; m < 2; m++)
#pragma unroll
    for (int t = 0; t < 4; t++)
      acc[m][t] = __builtin_amdgcn_wmma_f32_16x16x32_bf16(false, a[m].v, false, b[t],
                                                          (short)0, acc[m][t], false, false);
}

// =====================================================================
// bf16 WMMA GEMM: C[M x N] = A[M x K] (row-major bf16) * Bp (packed).
// Block = 128 rows x 64 cols; 4 waves share the same 64-col B stream
// (L0 reuse). Wave tile 32x64: 8 accumulators, double-buffered k-steps,
// B prefetched 4 k-steps ahead. Epilogues:
//  0: store bf16              1: +bias, exact GELU, store bf16
//  2: +bias +residual, f32    3: store f32 (guard col<N)
// =====================================================================
__global__ __launch_bounds__(128) void gemm_kernel(const __bf16* __restrict__ A,
                                                   const __bf16* __restrict__ Bp,
                                                   int N, int K, int lda, int ldc,
                                                   const float* __restrict__ bias,
                                                   const float* res,
                                                   float* outf, __bf16* outb, int epi) {
  int w = threadIdx.x >> 5, lane = threadIdx.x & 31;
  int hi = lane >> 4, lo = lane & 15;
  int col0 = blockIdx.x * 64;
  int rowW = blockIdx.y * 128 + w * 32;
  int KB = K >> 5;

  const __bf16* ar0 = A + (size_t)(rowW + lo) * lda;
  const __bf16* ar1 = A + (size_t)(rowW + 16 + lo) * lda;
  const __bf16* bb  = Bp + (size_t)(col0 >> 4) * (size_t)KB * 512 + lane * 16;

  v8f acc[2][4] = {};
  ABu a0[2], a1[2];
  v16bf b0[4], b1[4];

  load_frags(ar0, ar1, bb, KB, 0, hi, a0, b0);
  for (int kb = 0; kb < KB; kb += 2) {
    // prefetch B stream 4 k-steps ahead (each lane covers its 32B slice)
#pragma unroll
    for (int t = 0; t < 4; t++)
      __builtin_prefetch(bb + ((size_t)t * KB + kb + 4) * 512, 0, 1);
    load_frags(ar0, ar1, bb, KB, kb + 1, hi, a1, b1);
    wmma_all(acc, a0, b0);
    int kn = (kb + 2 < KB) ? kb + 2 : 0;   // dummy reload on last iter
    load_frags(ar0, ar1, bb, KB, kn, hi, a0, b0);
    wmma_all(acc, a1, b1);
  }

#pragma unroll
  for (int m = 0; m < 2; m++)
#pragma unroll
    for (int t = 0; t < 4; t++)
#pragma unroll
      for (int r = 0; r < 8; r++) {
        int row = rowW + m * 16 + r + 8 * hi;
        int col = col0 + t * 16 + lo;
        if (col < N) {
          float v = acc[m][t][r];
          size_t o = (size_t)row * ldc + col;
          if (epi == 0) {
            outb[o] = (__bf16)v;
          } else if (epi == 1) {
            float xx = v + bias[col];
            outb[o] = (__bf16)(0.5f * xx * (1.f + erff(xx * 0.70710678118f)));
          } else if (epi == 2) {
            outf[o] = v + bias[col] + res[o];
          } else {
            outf[o] = v;
          }
        }
      }
}

// =====================================================================
// Attention: one block per (b,h), 8 waves. T=128, D=256.
// S = q k^T * scale + mask (register softmax), O = P V via LDS-transposed V.
// Dynamic LDS: Vt[256][128] bf16 (64KB) + P[8][16][128] bf16 (32KB).
// =====================================================================
__global__ __launch_bounds__(256) void attention_kernel(const __bf16* __restrict__ q,
                                                        const __bf16* __restrict__ k,
                                                        const __bf16* __restrict__ v,
                                                        __bf16* __restrict__ o) {
  extern __shared__ char smem[];
  __bf16* Vt = (__bf16*)smem;                 // [DD][TT]
  __bf16* P  = Vt + (size_t)DD * TT;          // [8][16][TT]

  int bh = blockIdx.x;
  int bb = bh / HH, hh = bh % HH;
  const __bf16* qb = q + (size_t)bb * TT * HD + hh * DD;
  const __bf16* kb = k + (size_t)bb * TT * HD + hh * DD;
  const __bf16* vb = v + (size_t)bb * TT * HD + hh * DD;
  __bf16*       ob = o + (size_t)bb * TT * HD + hh * DD;

  int tid = threadIdx.x;
  for (int i = tid; i < TT * DD; i += 256) {
    int s = i / DD, d = i % DD;               // coalesced along d
    Vt[d * TT + s] = vb[(size_t)s * HD + d];
  }
  __syncthreads();

  int w = tid >> 5, lane = tid & 31;
  int hi = lane >> 4, lo = lane & 15;
  int t0 = w * 16;

  // ---- S = q k^T ----
  v8f acc[8] = {};
  for (int d0 = 0; d0 < DD; d0 += 32) {
    const __bf16* arow = qb + (size_t)(t0 + lo) * HD + d0;
    ABu af;
    af.h[0] = *(const v8bf*)(arow + hi * 16);
    af.h[1] = *(const v8bf*)(arow + hi * 16 + 8);
#pragma unroll
    for (int j = 0; j < 8; j++) {
      const __bf16* brow = kb + (size_t)(j * 16 + lo) * HD + d0 + hi * 16;
      ABu bfr;
      bfr.h[0] = *(const v8bf*)(brow);
      bfr.h[1] = *(const v8bf*)(brow + 8);
      acc[j] = __builtin_amdgcn_wmma_f32_16x16x32_bf16(false, af.v, false, bfr.v,
                                                       (short)0, acc[j], false, false);
    }
  }

  // ---- mask + softmax (per row, half-wave reductions) ----
  const float scale = 0.0625f;                // 1/sqrt(256)
#pragma unroll
  for (int r = 0; r < 8; r++) {
    int row = t0 + r + 8 * hi;
    float sv[8];
    float mx = -1e30f;
#pragma unroll
    for (int j = 0; j < 8; j++) {
      int col = j * 16 + lo;
      // faithful to reference: 0 where row < col, -1e9 otherwise
      float s = acc[j][r] * scale + ((row < col) ? 0.f : -1e9f);
      sv[j] = s; mx = fmaxf(mx, s);
    }
#pragma unroll
    for (int m = 8; m >= 1; m >>= 1) mx = fmaxf(mx, __shfl_xor(mx, m, 32));
    float sum = 0.f;
#pragma unroll
    for (int j = 0; j < 8; j++) { sv[j] = __expf(sv[j] - mx); sum += sv[j]; }
#pragma unroll
    for (int m = 8; m >= 1; m >>= 1) sum += __shfl_xor(sum, m, 32);
    float inv = 1.f / sum;
#pragma unroll
    for (int j = 0; j < 8; j++)
      P[(size_t)(t0 + r + 8 * hi) * TT + j * 16 + lo] = (__bf16)(sv[j] * inv);
  }
  // P written/read by the same wave only -> DScnt waits inserted by compiler.

  // ---- O = P @ V ----
  const __bf16* Pw = P + (size_t)t0 * TT;
  for (int dt = 0; dt < DD / 16; dt++) {
    v8f oc = {};
#pragma unroll
    for (int s0 = 0; s0 < TT; s0 += 32) {
      const __bf16* arow = Pw + (size_t)lo * TT + s0;
      ABu af;
      af.h[0] = *(const v8bf*)(arow + hi * 16);
      af.h[1] = *(const v8bf*)(arow + hi * 16 + 8);
      const __bf16* brow = Vt + (size_t)(dt * 16 + lo) * TT + s0 + hi * 16;
      ABu bfr;
      bfr.h[0] = *(const v8bf*)(brow);
      bfr.h[1] = *(const v8bf*)(brow + 8);
      oc = __builtin_amdgcn_wmma_f32_16x16x32_bf16(false, af.v, false, bfr.v,
                                                   (short)0, oc, false, false);
    }
#pragma unroll
    for (int r = 0; r < 8; r++)
      ob[(size_t)(t0 + r + 8 * hi) * HD + dt * 16 + lo] = (__bf16)oc[r];
  }
}

// =====================================================================
// Host orchestration
// =====================================================================
static inline int pack_blocks(int K, int Npad) {
  size_t total = (size_t)Npad * (size_t)K;
  return (int)((total + 255) / 256);
}

extern "C" void kernel_launch(void* const* d_in, const int* in_sizes, int n_in,
                              void* d_out, int out_size, void* d_ws, size_t ws_size,
                              hipStream_t stream) {
  const int*   x      = (const int*)  d_in[0];
  const float* wte    = (const float*)d_in[1];
  const float* wpe    = (const float*)d_in[2];
  const float* ln1_g  = (const float*)d_in[3];
  const float* ln1_b  = (const float*)d_in[4];
  const float* wq     = (const float*)d_in[5];
  const float* wk     = (const float*)d_in[6];
  const float* wv     = (const float*)d_in[7];
  const float* wo     = (const float*)d_in[8];
  const float* bo     = (const float*)d_in[9];
  const float* ln2_g  = (const float*)d_in[10];
  const float* ln2_b  = (const float*)d_in[11];
  const float* wfc    = (const float*)d_in[12];
  const float* bfc    = (const float*)d_in[13];
  const float* wproj  = (const float*)d_in[14];
  const float* bproj  = (const float*)d_in[15];
  const float* lnf_g  = (const float*)d_in[16];
  const float* lnf_b  = (const float*)d_in[17];
  const float* lmw    = (const float*)d_in[18];
  float* out = (float*)d_out;

  // ---- carve workspace (256B aligned slices; +1KB pad per packed buffer) ----
  char* base = (char*)d_ws;
  size_t off = 0;
  auto take = [&](size_t bytes) -> char* {
    char* p = base + off;
    off += (bytes + 1024 + 255) & ~(size_t)255;
    return p;
  };
  float*  hbuf = (float*) take((size_t)MM * EE * 4);
  __bf16* ybuf = (__bf16*)take((size_t)MM * EE * 2);
  __bf16* qb   = (__bf16*)take((size_t)MM * HD * 2);
  __bf16* kbf  = (__bf16*)take((size_t)MM * HD * 2);
  __bf16* vbf  = (__bf16*)take((size_t)MM * HD * 2);
  __bf16* obf  = (__bf16*)take((size_t)MM * HD * 2);
  __bf16* mid  = (__bf16*)take((size_t)MM * (4 * EE) * 2);
  __bf16* wqP  = (__bf16*)take((size_t)HD * EE * 2);
  __bf16* wkP  = (__bf16*)take((size_t)HD * EE * 2);
  __bf16* wvP  = (__bf16*)take((size_t)HD * EE * 2);
  __bf16* woP  = (__bf16*)take((size_t)EE * HD * 2);
  __bf16* wfP  = (__bf16*)take((size_t)(4 * EE) * EE * 2);
  __bf16* wpP  = (__bf16*)take((size_t)EE * (4 * EE) * 2);
  __bf16* lmP  = (__bf16*)take((size_t)NPAD_LM * EE * 2);
  (void)ws_size; (void)in_sizes; (void)n_in; (void)out_size;

  // ---- embedding ----
  embed_kernel<<<MM, 256, 0, stream>>>(x, wte, wpe, hbuf);

  const size_t smem_att = ((size_t)DD * TT + (size_t)8 * 16 * TT) * 2; // 96 KB

  for (int l = 0; l < LL; l++) {
    const float* Wq = wq + (size_t)l * EE * HD;
    const float* Wk = wk + (size_t)l * EE * HD;
    const float* Wv = wv + (size_t)l * EE * HD;
    const float* Wo = wo + (size_t)l * HD * EE;
    const float* Bo = bo + (size_t)l * EE;
    const float* Wf = wfc + (size_t)l * EE * 4 * EE;
    const float* Bf = bfc + (size_t)l * 4 * EE;
    const float* Wp = wproj + (size_t)l * 4 * EE * EE;
    const float* Bp = bproj + (size_t)l * EE;

    // LN1
    ln_kernel<<<MM / 8, 256, 0, stream>>>(hbuf, ln1_g + (size_t)l * EE,
                                          ln1_b + (size_t)l * EE, ybuf);
    // pack + QKV GEMMs  (K=256, N=2048)
    pack_kernel<<<pack_blocks(EE, HD), 256, 0, stream>>>(Wq, wqP, EE, HD, HD, 0);
    pack_kernel<<<pack_blocks(EE, HD), 256, 0, stream>>>(Wk, wkP, EE, HD, HD, 0);
    pack_kernel<<<pack_blocks(EE, HD), 256, 0, stream>>>(Wv, wvP, EE, HD, HD, 0);
    {
      dim3 g(HD / 64, MM / 128);
      gemm_kernel<<<g, 128, 0, stream>>>(ybuf, wqP, HD, EE, EE, HD, nullptr, nullptr, nullptr, qb, 0);
      gemm_kernel<<<g, 128, 0, stream>>>(ybuf, wkP, HD, EE, EE, HD, nullptr, nullptr, nullptr, kbf, 0);
      gemm_kernel<<<g, 128, 0, stream>>>(ybuf, wvP, HD, EE, EE, HD, nullptr, nullptr, nullptr, vbf, 0);
    }
    // attention
    attention_kernel<<<BB * HH, 256, smem_att, stream>>>(qb, kbf, vbf, obf);
    // output projection + bias + residual (K=2048, N=256)
    pack_kernel<<<pack_blocks(HD, EE), 256, 0, stream>>>(Wo, woP, HD, EE, EE, 0);
    {
      dim3 g(EE / 64, MM / 128);
      gemm_kernel<<<g, 128, 0, stream>>>(obf, woP, EE, HD, HD, EE, Bo, hbuf, hbuf, nullptr, 2);
    }
    // LN2
    ln_kernel<<<MM / 8, 256, 0, stream>>>(hbuf, ln2_g + (size_t)l * EE,
                                          ln2_b + (size_t)l * EE, ybuf);
    // MLP fc + GELU (K=256, N=1024)
    pack_kernel<<<pack_blocks(EE, 4 * EE), 256, 0, stream>>>(Wf, wfP, EE, 4 * EE, 4 * EE, 0);
    {
      dim3 g((4 * EE) / 64, MM / 128);
      gemm_kernel<<<g, 128, 0, stream>>>(ybuf, wfP, 4 * EE, EE, EE, 4 * EE, Bf, nullptr, nullptr, mid, 1);
    }
    // MLP proj + bias + residual (K=1024, N=256)
    pack_kernel<<<pack_blocks(4 * EE, EE), 256, 0, stream>>>(Wp, wpP, 4 * EE, EE, EE, 0);
    {
      dim3 g(EE / 64, MM / 128);
      gemm_kernel<<<g, 128, 0, stream>>>(mid, wpP, EE, 4 * EE, 4 * EE, EE, Bp, hbuf, hbuf, nullptr, 2);
    }
  }

  // final LN
  ln_kernel<<<MM / 8, 256, 0, stream>>>(hbuf, lnf_g, lnf_b, ybuf);
  // logits: pack lm_head^T (contiguous reads), GEMM K=256, N=50257 (Npad 50432)
  pack_kernel<<<pack_blocks(EE, NPAD_LM), 256, 0, stream>>>(lmw, lmP, EE, VOCAB, NPAD_LM, 1);
  {
    dim3 g(NPAD_LM / 64, MM / 128);
    gemm_kernel<<<g, 128, 0, stream>>>(ybuf, lmP, VOCAB, EE, EE, VOCAB, nullptr, nullptr, out, nullptr, 3);
  }
}
